// MashDecoder_61091614818585
// MI455X (gfx1250) — compile-verified, matrix operand order checked by reading, not run on hardware
//
#include <hip/hip_runtime.h>
#include <math.h>

// ---------------------------------------------------------------------------
// MashDecoder forward for gfx1250 (MI455X). All GEMMs run on the CDNA5 matrix
// path via v_wmma_f32_16x16x32_bf16 (bf16 A/B, f32 accumulate), software-
// pipelined so fragment loads for K-step k+32 are in flight during the WMMAs
// of step k. The selective-scan recurrence prefetches its B/C rows through
// the CDNA5 async global->LDS path (ASYNCcnt) with double buffering.
// ---------------------------------------------------------------------------

typedef __bf16 bf16_t;
typedef __attribute__((ext_vector_type(16))) __bf16 v16bf;
typedef __attribute__((ext_vector_type(8)))  __bf16 v8bf;
typedef __attribute__((ext_vector_type(8)))  float  v8f;

#define S_LEN   400
#define NBATCH  2
#define TOK     800           // B*S
#define DMODEL  400
#define DMODEL_P 416          // K padded to mult of 32
#define DINNER  800
#define DSTATE  128
#define NH      8
#define HD      100
#define NPROJ   1864
#define NPROJ_P 1920          // N padded to mult of 64
#define CONVD   1056
#define NLAYERS 96
#define MQ      2048
#define TQ      4096          // B*MQ
#define AH      4
#define ADH     400
#define ADH_P   416
#define ADH_R   448           // B-rows padded to mult of 64
#define NKV     3200
#define FF1     3200
#define FFH     1600

// -------------------------------- reductions -------------------------------
static __device__ __forceinline__ float warpSum(float v) {
#pragma unroll
  for (int off = 16; off > 0; off >>= 1) v += __shfl_down(v, off, 32);
  return v;
}
static __device__ __forceinline__ float warpMax(float v) {
#pragma unroll
  for (int off = 16; off > 0; off >>= 1) v = fmaxf(v, __shfl_down(v, off, 32));
  return v;
}
static __device__ __forceinline__ float blockSum(float v, float* sm) {
  v = warpSum(v);
  int nw = blockDim.x >> 5;
  if ((threadIdx.x & 31) == 0) sm[threadIdx.x >> 5] = v;
  __syncthreads();
  float t = 0.f;
  for (int i = 0; i < nw; ++i) t += sm[i];
  __syncthreads();
  return t;
}
static __device__ __forceinline__ float blockMax(float v, float* sm) {
  v = warpMax(v);
  int nw = blockDim.x >> 5;
  if ((threadIdx.x & 31) == 0) sm[threadIdx.x >> 5] = v;
  __syncthreads();
  float t = -3.4e38f;
  for (int i = 0; i < nw; ++i) t = fmaxf(t, sm[i]);
  __syncthreads();
  return t;
}

// ------------------------------ WMMA NT GEMM -------------------------------
// C[M,N](f32) = scale * A[M,K](bf16, lda) x B[N,K]^T(bf16, ldb) (+bias)(+=C)
// One wave per block; each wave computes a 16x64 tile (1 A-frag reused over
// 4 B-frags, 4 v_wmma_f32_16x16x32_bf16 per K-step). Software pipelined:
// fragments for step k+32 are loaded before the WMMAs of step k execute so
// vector loads overlap matrix ops. Requirements: M % 16 == 0, K % 32 == 0
// (staging buffers are K-zero-padded), B buffer rows padded to gridDim.y*64
// with zeros. Fragment layout per CDNA5 ISA 7.12.2 (16-bit A 16x32): lane =
// hi*16+row holds the row; halves K = k0+hi*8+[0..7], k0+16+hi*8+[0..7];
// B (32x16) mirrors it (NT symmetric).
static __device__ __forceinline__ v16bf load_frag(const bf16_t* p) {
  v8bf lo = *(const v8bf*)(p);
  v8bf hi = *(const v8bf*)(p + 16);
  return __builtin_shufflevector(lo, hi, 0, 1, 2, 3, 4, 5, 6, 7,
                                 8, 9, 10, 11, 12, 13, 14, 15);
}

__global__ __launch_bounds__(32)
void wmma_gemm_nt(const bf16_t* __restrict__ A, const bf16_t* __restrict__ B,
                  float* __restrict__ C, const float* __restrict__ bias,
                  int M, int N, int K, int lda, int ldb, int ldc,
                  long long sA, long long sB, long long sC,
                  float scale, int accum) {
  const int lane = threadIdx.x & 31;
  const int row  = lane & 15;
  const int hi   = lane >> 4;
  A += (long long)blockIdx.z * sA;
  B += (long long)blockIdx.z * sB;
  C += (long long)blockIdx.z * sC;
  const int tm = blockIdx.x * 16;
  const int tn = blockIdx.y * 64;

  v8f acc[4];
#pragma unroll
  for (int j = 0; j < 4; ++j)
#pragma unroll
    for (int r = 0; r < 8; ++r) acc[j][r] = 0.0f;

  const bf16_t* aPtr = A + (size_t)(tm + row) * lda + hi * 8;
  const bf16_t* bP0  = B + (size_t)(tn + row) * ldb + hi * 8;
  const bf16_t* bP1  = bP0 + (size_t)16 * ldb;
  const bf16_t* bP2  = bP0 + (size_t)32 * ldb;
  const bf16_t* bP3  = bP0 + (size_t)48 * ldb;

  // prologue: fragments for k=0
  v16bf aC  = load_frag(aPtr);
  v16bf bC0 = load_frag(bP0);
  v16bf bC1 = load_frag(bP1);
  v16bf bC2 = load_frag(bP2);
  v16bf bC3 = load_frag(bP3);

  for (int k0 = 0; k0 < K; k0 += 32) {
    // issue next K-step's loads before consuming current fragments
    const int kn = (k0 + 32 < K) ? (k0 + 32) : k0;   // last iter: redundant
    v16bf aN  = load_frag(aPtr + kn);
    v16bf bN0 = load_frag(bP0 + kn);
    v16bf bN1 = load_frag(bP1 + kn);
    v16bf bN2 = load_frag(bP2 + kn);
    v16bf bN3 = load_frag(bP3 + kn);
    __builtin_prefetch(aPtr + kn + 64, 0, 1);        // global_prefetch_b8

    acc[0] = __builtin_amdgcn_wmma_f32_16x16x32_bf16(
        false, aC, false, bC0, (short)0, acc[0], false, false);
    acc[1] = __builtin_amdgcn_wmma_f32_16x16x32_bf16(
        false, aC, false, bC1, (short)0, acc[1], false, false);
    acc[2] = __builtin_amdgcn_wmma_f32_16x16x32_bf16(
        false, aC, false, bC2, (short)0, acc[2], false, false);
    acc[3] = __builtin_amdgcn_wmma_f32_16x16x32_bf16(
        false, aC, false, bC3, (short)0, acc[3], false, false);

    aC = aN; bC0 = bN0; bC1 = bN1; bC2 = bN2; bC3 = bN3;
  }

#pragma unroll
  for (int j = 0; j < 4; ++j) {
    int n = tn + j * 16 + row;
    if (n >= N) continue;
    float bv = bias ? bias[n] : 0.0f;
#pragma unroll
    for (int r = 0; r < 8; ++r) {
      int m = tm + hi * 8 + r;
      if (m >= M) continue;
      size_t idx = (size_t)m * ldc + n;
      float v = acc[j][r] * scale + bv;
      if (accum) v += C[idx];
      C[idx] = v;
    }
  }
}

// --------------------------- conversion / padding --------------------------
__global__ void cvt_pad_kernel(const float* __restrict__ src, int srcLD,
                               bf16_t* __restrict__ dst, int dstLD,
                               int rows, int cols, int rowsPad) {
  size_t idx = (size_t)blockIdx.x * blockDim.x + threadIdx.x;
  size_t total = (size_t)rowsPad * dstLD;
  if (idx >= total) return;
  int c = (int)(idx % dstLD);
  int r = (int)(idx / dstLD);
  float v = (r < rows && c < cols) ? src[(size_t)r * srcLD + c] : 0.0f;
  dst[idx] = (bf16_t)v;
}

// ------------------------------- point embed -------------------------------
__global__ void point_embed_kernel(const float* __restrict__ x, int xld,
                                   int xdim, int nfreq,
                                   const float* __restrict__ W,
                                   const float* __restrict__ bias,
                                   float* __restrict__ out, int outLD,
                                   int outOff, int nOut, int nTok) {
  int idx = blockIdx.x * blockDim.x + threadIdx.x;
  if (idx >= nTok * nOut) return;
  int o = idx % nOut;
  int t = idx / nOut;
  int embDim = xdim * nfreq * 2 + xdim;
  const float* wr = W + (size_t)o * embDim;
  float acc = bias[o];
  for (int d = 0; d < xdim; ++d) {
    float xv = x[(size_t)t * xld + d];
    float f = 3.14159265358979323846f;
    for (int k = 0; k < nfreq; ++k) {
      float ph = xv * f;
      acc += sinf(ph) * wr[d * nfreq + k] +
             cosf(ph) * wr[xdim * nfreq + d * nfreq + k];
      f *= 2.0f;
    }
    acc += xv * wr[2 * xdim * nfreq + d];
  }
  out[(size_t)t * outLD + outOff + o] = acc;
}

// ------------------------ res += h ; rmsnorm -> bf16 -----------------------
__global__ __launch_bounds__(128)
void add_rmsnorm_kernel(const float* __restrict__ hs, float* __restrict__ res,
                        const float* __restrict__ w, bf16_t* outbf,
                        float* outf) {
  __shared__ float sm[8];
  int t = blockIdx.x;
  float vals[4];
  float ss = 0.f;
  int it = 0;
  for (int c = threadIdx.x; c < DMODEL; c += 128, ++it) {
    float v = res[(size_t)t * DMODEL + c] + hs[(size_t)t * DMODEL + c];
    res[(size_t)t * DMODEL + c] = v;
    vals[it] = v;
    ss += v * v;
  }
  float tot = blockSum(ss, sm);
  float r = rsqrtf(tot / DMODEL + 1e-5f);
  it = 0;
  for (int c = threadIdx.x; c < DMODEL; c += 128, ++it) {
    float o = vals[it] * r * w[c];
    if (outbf) outbf[(size_t)t * DMODEL_P + c] = (bf16_t)o;
    if (outf)  outf[(size_t)t * DMODEL + c] = o;
  }
  if (outbf)
    for (int c = DMODEL + threadIdx.x; c < DMODEL_P; c += 128)
      outbf[(size_t)t * DMODEL_P + c] = (bf16_t)0.0f;
}

// ------------------------------ layernorm -> bf16 --------------------------
__global__ __launch_bounds__(128)
void layernorm_kernel(const float* __restrict__ x, const float* __restrict__ g,
                      const float* __restrict__ b, bf16_t* __restrict__ outbf,
                      int C, int Cpad) {
  __shared__ float sm[8];
  int t = blockIdx.x;
  float vals[4];
  float s1 = 0.f;
  int it = 0;
  for (int c = threadIdx.x; c < C; c += 128, ++it) {
    float v = x[(size_t)t * C + c];
    vals[it] = v;
    s1 += v;
  }
  float mu = blockSum(s1, sm) / C;
  float s2 = 0.f;
  it = 0;
  for (int c = threadIdx.x; c < C; c += 128, ++it) {
    float d = vals[it] - mu;
    s2 += d * d;
  }
  float var = blockSum(s2, sm) / C;
  float r = rsqrtf(var + 1e-5f);
  it = 0;
  for (int c = threadIdx.x; c < C; c += 128, ++it)
    outbf[(size_t)t * Cpad + c] = (bf16_t)((vals[it] - mu) * r * g[c] + b[c]);
  for (int c = C + threadIdx.x; c < Cpad; c += 128)
    outbf[(size_t)t * Cpad + c] = (bf16_t)0.0f;
}

// ------------------------- causal depthwise conv + SiLU --------------------
__global__ void conv_silu_kernel(const float* __restrict__ zx,
                                 const float* __restrict__ cw,
                                 const float* __restrict__ cb,
                                 float* __restrict__ xBC) {
  int idx = blockIdx.x * blockDim.x + threadIdx.x;
  if (idx >= TOK * CONVD) return;
  int c = idx % CONVD;
  int t = idx / CONVD;
  int s = t % S_LEN;
  int b = t / S_LEN;
  float acc = cb[c];
#pragma unroll
  for (int k = 0; k < 4; ++k) {
    int sp = s + k - 3;
    if (sp >= 0)
      acc += zx[((size_t)(b * S_LEN + sp)) * NPROJ_P + DINNER + c] * cw[c * 4 + k];
  }
  xBC[idx] = acc / (1.0f + __expf(-acc));
}

// ---------------------------- dt softplus + dA -----------------------------
__global__ void dt_kernel(const float* __restrict__ zx,
                          const float* __restrict__ dt_bias,
                          const float* __restrict__ A_log,
                          float* __restrict__ dtb, float* __restrict__ dAb) {
  int idx = blockIdx.x * blockDim.x + threadIdx.x;
  if (idx >= TOK * NH) return;
  int h = idx % NH;
  int t = idx / NH;
  float x = zx[(size_t)t * NPROJ_P + (NPROJ - NH) + h] + dt_bias[h];
  float dt = (x > 20.0f) ? x : log1pf(__expf(x));
  float A = -__expf(A_log[h]);
  dtb[idx] = dt;
  dAb[idx] = __expf(dt * A);
}

// ----------------------------- selective scan ------------------------------
// block = (b,h,p); thread n owns one state element; per-step 128-wide C-dot.
// B/C rows for step s+1 stream into an LDS double buffer through the CDNA5
// async global->LDS path (ASYNCcnt) while step s computes; async loads return
// in order, so "s_wait_asynccnt 2" = the current phase's pair has landed.
// LDS byte address = low 32 bits of the flat pointer (LDS aperture mapping,
// ISA 10.2).
__global__ __launch_bounds__(128)
void scan_kernel(const float* __restrict__ xBC, const float* __restrict__ dtb,
                 const float* __restrict__ dAb, float* __restrict__ y) {
  __shared__ float partial[4];
  __shared__ float sbuf[2][2][DSTATE];   // [phase][B|C][n]
  int bid = blockIdx.x;          // 0..1599
  int p = bid % HD;
  int h = (bid / HD) % NH;
  int b = bid / (HD * NH);
  int n = threadIdx.x;           // 0..127
  unsigned ldsB[2], ldsC[2];
  ldsB[0] = (unsigned)(uintptr_t)&sbuf[0][0][n];
  ldsC[0] = (unsigned)(uintptr_t)&sbuf[0][1][n];
  ldsB[1] = (unsigned)(uintptr_t)&sbuf[1][0][n];
  ldsC[1] = (unsigned)(uintptr_t)&sbuf[1][1][n];

  auto prefetch = [&](int s, int phase) {
    const float* gB = xBC + (size_t)(b * S_LEN + s) * CONVD + DINNER + n;
    const float* gC = gB + DSTATE;
    asm volatile("global_load_async_to_lds_b32 %0, %1, off"
                 :: "v"(ldsB[phase]), "v"(gB) : "memory");
    asm volatile("global_load_async_to_lds_b32 %0, %1, off"
                 :: "v"(ldsC[phase]), "v"(gC) : "memory");
  };

  prefetch(0, 0);
  float state = 0.0f;
  int ph = 0;
  for (int s = 0; s < S_LEN; ++s) {
    int t = b * S_LEN + s;
    if (s + 1 < S_LEN) {
      prefetch(s + 1, ph ^ 1);
      asm volatile("s_wait_asynccnt 0x2" ::: "memory");
    } else {
      asm volatile("s_wait_asynccnt 0x0" ::: "memory");
    }
    float dAv = dAb[t * NH + h];
    float dtv = dtb[t * NH + h];
    float xv  = xBC[(size_t)t * CONVD + h * HD + p];
    float Bv  = sbuf[ph][0][n];
    float Cv  = sbuf[ph][1][n];
    state = state * dAv + dtv * xv * Bv;
    float v = warpSum(state * Cv);
    if ((threadIdx.x & 31) == 0) partial[threadIdx.x >> 5] = v;
    __syncthreads();
    if (threadIdx.x == 0)
      y[(size_t)t * DINNER + h * HD + p] =
          partial[0] + partial[1] + partial[2] + partial[3];
    __syncthreads();
    ph ^= 1;
  }
}

// ---------------- y + D*x, gate with silu(z), gated RMSNorm -> bf16 --------
__global__ __launch_bounds__(256)
void gate_norm_kernel(const float* __restrict__ y, const float* __restrict__ xBC,
                      const float* __restrict__ zx, const float* __restrict__ Dp,
                      const float* __restrict__ gw, bf16_t* __restrict__ out) {
  __shared__ float sm[8];
  int t = blockIdx.x;
  float vals[4];
  float ss = 0.f;
  int it = 0;
  for (int c = threadIdx.x; c < DINNER; c += 256, ++it) {
    int h = c / HD;
    float yy = y[(size_t)t * DINNER + c] + Dp[h] * xBC[(size_t)t * CONVD + c];
    float z = zx[(size_t)t * NPROJ_P + c];
    float g = yy * (z / (1.0f + __expf(-z)));
    vals[it] = g;
    ss += g * g;
  }
  float tot = blockSum(ss, sm);
  float r = rsqrtf(tot / DINNER + 1e-5f);
  it = 0;
  for (int c = threadIdx.x; c < DINNER; c += 256, ++it)
    out[(size_t)t * DINNER + c] = (bf16_t)(vals[it] * r * gw[c]);
}

// ----------------------------- attention helpers ---------------------------
__global__ void split_heads_q_kernel(const float* __restrict__ q,
                                     bf16_t* __restrict__ qbf) {
  size_t idx = (size_t)blockIdx.x * blockDim.x + threadIdx.x;
  size_t total = (size_t)NBATCH * AH * MQ * ADH_P;
  if (idx >= total) return;
  int d = (int)(idx % ADH_P);
  int m = (int)((idx / ADH_P) % MQ);
  int bh = (int)(idx / ((size_t)ADH_P * MQ));
  int b = bh / AH, h = bh % AH;
  float v = (d < ADH) ? q[((size_t)(b * MQ + m)) * (AH * ADH) + h * ADH + d] : 0.f;
  qbf[idx] = (bf16_t)v;
}

__global__ void make_kbf_kernel(const float* __restrict__ kv,
                                bf16_t* __restrict__ kbf) {
  size_t idx = (size_t)blockIdx.x * blockDim.x + threadIdx.x;
  size_t total = (size_t)NBATCH * AH * ADH_R * ADH_P;
  if (idx >= total) return;
  int d = (int)(idx % ADH_P);
  int n = (int)((idx / ADH_P) % ADH_R);
  int bh = (int)(idx / ((size_t)ADH_P * ADH_R));
  int b = bh / AH, h = bh % AH;
  float v = (n < S_LEN && d < ADH)
                ? kv[((size_t)(b * S_LEN + n)) * NKV + h * ADH + d] : 0.f;
  kbf[idx] = (bf16_t)v;
}

__global__ void make_vt_kernel(const float* __restrict__ kv,
                               bf16_t* __restrict__ vt) {
  size_t idx = (size_t)blockIdx.x * blockDim.x + threadIdx.x;
  size_t total = (size_t)NBATCH * AH * ADH_R * ADH_P;
  if (idx >= total) return;
  int n = (int)(idx % ADH_P);
  int d = (int)((idx / ADH_P) % ADH_R);
  int bh = (int)(idx / ((size_t)ADH_P * ADH_R));
  int b = bh / AH, h = bh % AH;
  float v = (d < ADH && n < S_LEN)
                ? kv[((size_t)(b * S_LEN + n)) * NKV + AH * ADH + h * ADH + d] : 0.f;
  vt[idx] = (bf16_t)v;
}

__global__ __launch_bounds__(128)
void softmax_kernel(const float* __restrict__ scores, bf16_t* __restrict__ attn) {
  __shared__ float sm[8];
  int row = blockIdx.x;                       // bh*MQ + m, 16384 rows
  const float* sp = scores + (size_t)row * ADH;
  float vals[4];
  int it = 0;
  float mx = -3.4e38f;
  for (int c = threadIdx.x; c < ADH; c += 128, ++it) {
    vals[it] = sp[c];
    mx = fmaxf(mx, vals[it]);
  }
  mx = blockMax(mx, sm);
  float ss = 0.f;
  it = 0;
  for (int c = threadIdx.x; c < ADH; c += 128, ++it) {
    vals[it] = __expf(vals[it] - mx);
    ss += vals[it];
  }
  float inv = 1.0f / blockSum(ss, sm);
  it = 0;
  for (int c = threadIdx.x; c < ADH; c += 128, ++it)
    attn[(size_t)row * ADH_P + c] = (bf16_t)(vals[it] * inv);
  for (int c = ADH + threadIdx.x; c < ADH_P; c += 128)
    attn[(size_t)row * ADH_P + c] = (bf16_t)0.0f;
}

__global__ void merge_heads_kernel(const float* __restrict__ o2,
                                   bf16_t* __restrict__ obf) {
  size_t idx = (size_t)blockIdx.x * blockDim.x + threadIdx.x;
  size_t total = (size_t)TQ * (AH * ADH);
  if (idx >= total) return;
  int c = (int)(idx % (AH * ADH));
  size_t mrow = idx / (AH * ADH);
  int b = (int)(mrow / MQ);
  int m = (int)(mrow % MQ);
  int h = c / ADH, d = c % ADH;
  obf[idx] = (bf16_t)o2[(((size_t)(b * AH + h)) * MQ + m) * ADH + d];
}

// -------------------------------- GeGLU ------------------------------------
__global__ void geglu_kernel(const float* __restrict__ fx,
                             bf16_t* __restrict__ gg) {
  size_t idx = (size_t)blockIdx.x * blockDim.x + threadIdx.x;
  size_t total = (size_t)TQ * FFH;
  if (idx >= total) return;
  size_t t = idx / FFH;
  int j = (int)(idx % FFH);
  float u = fx[t * FF1 + j];
  float g = fx[t * FF1 + FFH + j];
  float ge = 0.5f * g * (1.0f + erff(g * 0.70710678118654752f));
  gg[idx] = (bf16_t)(u * ge);
}

// ------------------------------- occupancy head ----------------------------
__global__ void occ_kernel(const float* __restrict__ lat,
                           const float* __restrict__ w,
                           const float* __restrict__ b, float* __restrict__ out) {
  int m = blockIdx.x * blockDim.x + threadIdx.x;
  if (m >= TQ) return;
  float acc = b[0];
  for (int c = 0; c < DMODEL; ++c) acc += lat[(size_t)m * DMODEL + c] * w[c];
  out[m] = acc;
}

// ===========================================================================
extern "C" void kernel_launch(void* const* d_in, const int* in_sizes, int n_in,
                              void* d_out, int out_size, void* d_ws,
                              size_t ws_size, hipStream_t stream) {
  (void)in_sizes; (void)n_in; (void)out_size; (void)ws_size;
  // ---- inputs, in setup_inputs() flattened (depth-first, insertion) order
  const float* mash   = (const float*)d_in[0];
  const float* qry    = (const float*)d_in[1];
  const float* rot_W  = (const float*)d_in[2];
  const float* rot_b  = (const float*)d_in[3];
  const float* pos_W  = (const float*)d_in[4];
  const float* pos_b  = (const float*)d_in[5];
  const float* mask_W = (const float*)d_in[6];
  const float* mask_b = (const float*)d_in[7];
  const float* sh_W   = (const float*)d_in[8];
  const float* sh_b   = (const float*)d_in[9];
  const float* point_W= (const float*)d_in[10];
  const float* point_b= (const float*)d_in[11];
  const float* norm_w = (const float*)d_in[12];
  const float* in_w   = (const float*)d_in[13];
  const float* conv_w = (const float*)d_in[14];
  const float* conv_b = (const float*)d_in[15];
  const float* dt_bias= (const float*)d_in[16];
  const float* A_log  = (const float*)d_in[17];
  const float* Dpar   = (const float*)d_in[18];
  const float* gnorm_w= (const float*)d_in[19];
  const float* out_w  = (const float*)d_in[20];
  const float* norm_f = (const float*)d_in[21];
  const float* ln_q_g = (const float*)d_in[22];
  const float* ln_q_b = (const float*)d_in[23];
  const float* ln_c_g = (const float*)d_in[24];
  const float* ln_c_b = (const float*)d_in[25];
  const float* Wq     = (const float*)d_in[26];
  const float* Wkv    = (const float*)d_in[27];
  const float* Wo     = (const float*)d_in[28];
  const float* bo     = (const float*)d_in[29];
  const float* ln_g   = (const float*)d_in[30];
  const float* ln_b   = (const float*)d_in[31];
  const float* W1     = (const float*)d_in[32];
  const float* b1     = (const float*)d_in[33];
  const float* W2     = (const float*)d_in[34];
  const float* b2     = (const float*)d_in[35];
  const float* outW   = (const float*)d_in[36];
  const float* outB   = (const float*)d_in[37];
  float* occ = (float*)d_out;
  char* ws = (char*)d_ws;

  // ---- workspace layout (~120 MB); arena aliases phase-A / phase-B scratch
  size_t cur = 0;
  auto take = [&](size_t bytes) {
    cur = (cur + 255) & ~(size_t)255;
    size_t o = cur; cur += bytes; return o;
  };
  const size_t o_h    = take((size_t)TOK * DMODEL * 4);
  const size_t o_res  = take((size_t)TOK * DMODEL * 4);
  const size_t o_lat  = take((size_t)TQ * DMODEL * 4);
  const size_t o_wqb  = take((size_t)1600 * DMODEL_P * 2);
  const size_t o_wkvb = take((size_t)3200 * DMODEL_P * 2);
  const size_t o_wob  = take((size_t)ADH_R * 1600 * 2);
  const size_t o_w1b  = take((size_t)3200 * DMODEL_P * 2);
  const size_t o_w2b  = take((size_t)ADH_R * 1600 * 2);
  cur = (cur + 255) & ~(size_t)255;
  const size_t arena = cur;

  // phase A (per-layer scratch)
  size_t a = arena;
  auto takeA = [&](size_t bytes) {
    a = (a + 255) & ~(size_t)255;
    size_t o = a; a += bytes; return o;
  };
  const size_t o_xn   = takeA((size_t)TOK * DMODEL_P * 2);
  const size_t o_win  = takeA((size_t)NPROJ_P * DMODEL_P * 2);
  const size_t o_wout = takeA((size_t)ADH_R * DINNER * 2);
  const size_t o_zx   = takeA((size_t)TOK * NPROJ_P * 4);
  const size_t o_xbc  = takeA((size_t)TOK * CONVD * 4);
  const size_t o_dt   = takeA((size_t)TOK * NH * 4);
  const size_t o_dA   = takeA((size_t)TOK * NH * 4);
  const size_t o_y    = takeA((size_t)TOK * DINNER * 4);
  const size_t o_gate = takeA((size_t)TOK * DINNER * 2);

  // phase B (aliases phase-A scratch; all phase-A buffers dead by then)
  size_t bcur = arena;
  auto takeB = [&](size_t bytes) {
    bcur = (bcur + 255) & ~(size_t)255;
    size_t o = bcur; bcur += bytes; return o;
  };
  const size_t o_qemb = takeB((size_t)TQ * DMODEL * 4);
  const size_t o_qn   = takeB((size_t)TQ * DMODEL_P * 2);
  const size_t o_cn   = takeB((size_t)TOK * DMODEL_P * 2);
  const size_t o_kv   = takeB((size_t)TOK * NKV * 4);
  const size_t o_kbf  = takeB((size_t)NBATCH * AH * ADH_R * ADH_P * 2);
  const size_t o_vt   = takeB((size_t)NBATCH * AH * ADH_R * ADH_P * 2);
  const size_t o_qbf  = takeB((size_t)NBATCH * AH * MQ * ADH_P * 2);
  const size_t o_attn = takeB((size_t)NBATCH * AH * MQ * ADH_P * 2); // also obf
  const size_t o_big  = takeB((size_t)TQ * 1600 * 4);  // qf32 -> scores -> o2
  const size_t o_fn   = takeB((size_t)TQ * DMODEL_P * 2);
  const size_t o_gg   = takeB((size_t)TQ * FFH * 2);
  const size_t o_fx   = arena;  // 4096x3200 f32 overlaps dead early buffers

  auto F  = [&](size_t o) { return (float*)(ws + o); };
  auto BF = [&](size_t o) { return (bf16_t*)(ws + o); };

  auto gemm = [&](const bf16_t* A, const bf16_t* Bm, float* C,
                  const float* bias, int M, int N, int K, int lda, int ldb,
                  int ldc, long long sA, long long sB, long long sC,
                  int batches, float scale, int accum) {
    dim3 g(M / 16, (N + 63) / 64, batches);
    wmma_gemm_nt<<<g, 32, 0, stream>>>(A, Bm, C, bias, M, N, K, lda, ldb, ldc,
                                       sA, sB, sC, scale, accum);
  };
  auto cvt = [&](const float* src, int srcLD, bf16_t* dst, int dstLD, int rows,
                 int cols, int rowsPad) {
    size_t total = (size_t)rowsPad * dstLD;
    cvt_pad_kernel<<<(unsigned)((total + 255) / 256), 256, 0, stream>>>(
        src, srcLD, dst, dstLD, rows, cols, rowsPad);
  };

  // ---------------- embeddings: h = concat(point_embed x4) ----------------
  hipMemsetAsync(ws + o_res, 0, (size_t)TOK * DMODEL * 4, stream);
  auto pe = [&](const float* x, int xdim, int nfreq, const float* W,
                const float* bias, float* out, int outOff, int nOut, int nTok,
                int xld) {
    int tot = nTok * nOut;
    point_embed_kernel<<<(tot + 255) / 256, 256, 0, stream>>>(
        x, xld, xdim, nfreq, W, bias, out, DMODEL, outOff, nOut, nTok);
  };
  pe(mash + 0, 6, 4, rot_W, rot_b, F(o_h), 0, 100, TOK, 25);
  pe(mash + 6, 3, 8, pos_W, pos_b, F(o_h), 100, 100, TOK, 25);
  pe(mash + 9, 7, 3, mask_W, mask_b, F(o_h), 200, 100, TOK, 25);
  pe(mash + 16, 9, 2, sh_W, sh_b, F(o_h), 300, 100, TOK, 25);

  // ---------------------------- 96 Mamba2 layers --------------------------
  for (int L = 0; L < NLAYERS; ++L) {
    const float* nw   = norm_w + (size_t)L * DMODEL;
    const float* iw   = in_w + (size_t)L * NPROJ * DMODEL;
    const float* cw   = conv_w + (size_t)L * CONVD * 4;
    const float* cb   = conv_b + (size_t)L * CONVD;
    const float* dtbp = dt_bias + (size_t)L * NH;
    const float* alp  = A_log + (size_t)L * NH;
    const float* dp   = Dpar + (size_t)L * NH;
    const float* gwp  = gnorm_w + (size_t)L * DINNER;
    const float* ow   = out_w + (size_t)L * DMODEL * DINNER;

    // res += h ; xnorm = rmsnorm(res)*norm_w -> bf16 (K-padded)
    add_rmsnorm_kernel<<<TOK, 128, 0, stream>>>(F(o_h), F(o_res), nw,
                                                BF(o_xn), nullptr);
    // in_proj (WMMA): zx = xnorm @ in_w^T
    cvt(iw, DMODEL, BF(o_win), DMODEL_P, NPROJ, DMODEL, NPROJ_P);
    gemm(BF(o_xn), BF(o_win), F(o_zx), nullptr, TOK, NPROJ, DMODEL_P,
         DMODEL_P, DMODEL_P, NPROJ_P, 0, 0, 0, 1, 1.0f, 0);
    // conv + SiLU, dt/dA
    conv_silu_kernel<<<(TOK * CONVD + 255) / 256, 256, 0, stream>>>(
        F(o_zx), cw, cb, F(o_xbc));
    dt_kernel<<<(TOK * NH + 255) / 256, 256, 0, stream>>>(F(o_zx), dtbp, alp,
                                                          F(o_dt), F(o_dA));
    // selective scan (async LDS double-buffered B/C prefetch)
    scan_kernel<<<NBATCH * NH * HD, 128, 0, stream>>>(F(o_xbc), F(o_dt),
                                                      F(o_dA), F(o_y));
    // +D*x, gate, gated RMSNorm -> bf16
    gate_norm_kernel<<<TOK, 256, 0, stream>>>(F(o_y), F(o_xbc), F(o_zx), dp,
                                              gwp, BF(o_gate));
    // out_proj (WMMA): h = gated @ out_w^T
    cvt(ow, DINNER, BF(o_wout), DINNER, DMODEL, DINNER, ADH_R);
    gemm(BF(o_gate), BF(o_wout), F(o_h), nullptr, TOK, DMODEL, DINNER, DINNER,
         DINNER, DMODEL, 0, 0, 0, 1, 1.0f, 0);
  }
  // final fused add + rmsnorm -> h (f32)
  add_rmsnorm_kernel<<<TOK, 128, 0, stream>>>(F(o_h), F(o_res), norm_f,
                                              nullptr, F(o_h));

  // ----------------------------- cross attention --------------------------
  pe(qry, 3, 8, point_W, point_b, F(o_qemb), 0, DMODEL, TQ, 3);
  layernorm_kernel<<<TQ, 128, 0, stream>>>(F(o_qemb), ln_q_g, ln_q_b,
                                           BF(o_qn), DMODEL, DMODEL_P);
  layernorm_kernel<<<TOK, 128, 0, stream>>>(F(o_h), ln_c_g, ln_c_b, BF(o_cn),
                                            DMODEL, DMODEL_P);
  cvt(Wq, DMODEL, BF(o_wqb), DMODEL_P, 1600, DMODEL, 1600);
  cvt(Wkv, DMODEL, BF(o_wkvb), DMODEL_P, 3200, DMODEL, 3200);
  cvt(Wo, 1600, BF(o_wob), 1600, DMODEL, 1600, ADH_R);
  cvt(W1, DMODEL, BF(o_w1b), DMODEL_P, 3200, DMODEL, 3200);
  cvt(W2, FFH, BF(o_w2b), FFH, DMODEL, FFH, ADH_R);

  // q = qn @ Wq^T ; kv = cn @ Wkv^T
  gemm(BF(o_qn), BF(o_wqb), F(o_big), nullptr, TQ, 1600, DMODEL_P, DMODEL_P,
       DMODEL_P, 1600, 0, 0, 0, 1, 1.0f, 0);
  gemm(BF(o_cn), BF(o_wkvb), F(o_kv), nullptr, TOK, NKV, DMODEL_P, DMODEL_P,
       DMODEL_P, NKV, 0, 0, 0, 1, 1.0f, 0);
  {
    size_t tq = (size_t)NBATCH * AH * MQ * ADH_P;
    split_heads_q_kernel<<<(unsigned)((tq + 255) / 256), 256, 0, stream>>>(
        F(o_big), BF(o_qbf));
    size_t tk = (size_t)NBATCH * AH * ADH_R * ADH_P;
    make_kbf_kernel<<<(unsigned)((tk + 255) / 256), 256, 0, stream>>>(
        F(o_kv), BF(o_kbf));
    make_vt_kernel<<<(unsigned)((tk + 255) / 256), 256, 0, stream>>>(
        F(o_kv), BF(o_vt));
  }
  // scores = q k^T / sqrt(dH)   (batched over b*h; overwrites dead qf32)
  gemm(BF(o_qbf), BF(o_kbf), F(o_big), nullptr, MQ, ADH, ADH_P, ADH_P, ADH_P,
       ADH, (long long)MQ * ADH_P, (long long)ADH_R * ADH_P,
       (long long)MQ * ADH, NBATCH * AH, 0.05f, 0);
  softmax_kernel<<<NBATCH * AH * MQ, 128, 0, stream>>>(F(o_big), BF(o_attn));
  // o = attn @ v   (batched; overwrites dead scores)
  gemm(BF(o_attn), BF(o_vt), F(o_big), nullptr, MQ, ADH, ADH_P, ADH_P, ADH_P,
       ADH, (long long)MQ * ADH_P, (long long)ADH_R * ADH_P,
       (long long)MQ * ADH, NBATCH * AH, 1.0f, 0);
  {
    size_t tot = (size_t)TQ * (AH * ADH);   // obf reuses attn buffer (dead)
    merge_heads_kernel<<<(unsigned)((tot + 255) / 256), 256, 0, stream>>>(
        F(o_big), BF(o_attn));
  }
  // latents = o @ Wo^T + bo
  gemm(BF(o_attn), BF(o_wob), F(o_lat), bo, TQ, DMODEL, 1600, 1600, 1600,
       DMODEL, 0, 0, 0, 1, 1.0f, 0);

  // -------------------------------- GeGLU FF -------------------------------
  layernorm_kernel<<<TQ, 128, 0, stream>>>(F(o_lat), ln_g, ln_b, BF(o_fn),
                                           DMODEL, DMODEL_P);
  gemm(BF(o_fn), BF(o_w1b), F(o_fx), b1, TQ, FF1, DMODEL_P, DMODEL_P,
       DMODEL_P, FF1, 0, 0, 0, 1, 1.0f, 0);
  {
    size_t tot = (size_t)TQ * FFH;
    geglu_kernel<<<(unsigned)((tot + 255) / 256), 256, 0, stream>>>(F(o_fx),
                                                                    BF(o_gg));
  }
  gemm(BF(o_gg), BF(o_w2b), F(o_lat), b2, TQ, DMODEL, FFH, FFH, FFH, DMODEL,
       0, 0, 0, 1, 1.0f, 1);   // latents += ... + b2

  // --------------------------------- head ---------------------------------
  occ_kernel<<<(TQ + 255) / 256, 256, 0, stream>>>(F(o_lat), outW, outB, occ);
}
